// TransformerBlock_28544352649221
// MI455X (gfx1250) — matrix operand order repaired and checked by reading, not compiled, not run on hardware
//
#include <hip/hip_runtime.h>

// ---------------------------------------------------------------------------
// CDNA5 (gfx1250) types for WMMA bf16
// ---------------------------------------------------------------------------
typedef __attribute__((ext_vector_type(16))) __bf16 v16bf;
typedef __attribute__((ext_vector_type(8)))  float  v8f;

union BF16x16 {
    v16bf  v;
    uint4  u[2];
    __bf16 e[16];
};

// Raw 32-bit LDS byte offset of a generic pointer into shared memory
// (addrspace(3) pointers on amdgcn are LDS offsets).
__device__ inline unsigned lds_offset(const void* p)
{
    return (unsigned)(size_t)(__attribute__((address_space(3))) const void*)p;
}

// ---------------------------------------------------------------------------
// LayerNorm (C==1024) + cast to bf16.  One row per 256-thread block.
// ---------------------------------------------------------------------------
__global__ __launch_bounds__(256)
void layernorm_bf16(const float* __restrict__ x,
                    const float* __restrict__ w,
                    const float* __restrict__ b,
                    __bf16* __restrict__ out)
{
    __shared__ float red[256];
    const int row = blockIdx.x;
    const int t   = threadIdx.x;
    const float* xr = x + (long)row * 1024;

    float4 xv = ((const float4*)xr)[t];
    float s = xv.x + xv.y + xv.z + xv.w;
    red[t] = s; __syncthreads();
    #pragma unroll
    for (int o = 128; o > 0; o >>= 1) { if (t < o) red[t] += red[t + o]; __syncthreads(); }
    const float mu = red[0] * (1.0f / 1024.0f);
    __syncthreads();

    float dx = xv.x - mu, dy = xv.y - mu, dz = xv.z - mu, dw = xv.w - mu;
    red[t] = dx*dx + dy*dy + dz*dz + dw*dw; __syncthreads();
    #pragma unroll
    for (int o = 128; o > 0; o >>= 1) { if (t < o) red[t] += red[t + o]; __syncthreads(); }
    const float rstd = rsqrtf(red[0] * (1.0f / 1024.0f) + 1e-5f);

    float4 wv = ((const float4*)w)[t];
    float4 bv = ((const float4*)b)[t];
    __bf16* orow = out + (long)row * 1024 + t * 4;
    orow[0] = (__bf16)(dx * rstd * wv.x + bv.x);
    orow[1] = (__bf16)(dy * rstd * wv.y + bv.y);
    orow[2] = (__bf16)(dz * rstd * wv.z + bv.z);
    orow[3] = (__bf16)(dw * rstd * wv.w + bv.w);
}

// ---------------------------------------------------------------------------
// Plain f32 -> bf16 cast (activations and row-major (K,N) weights)
// ---------------------------------------------------------------------------
__global__ __launch_bounds__(256)
void cast_f32_to_bf16(const float* __restrict__ in, __bf16* __restrict__ out, long n)
{
    long i = (long)blockIdx.x * 256 + threadIdx.x;
    if (i < n) out[i] = (__bf16)in[i];
}

// ---------------------------------------------------------------------------
// Permute per-head einsum weight (H, K, hb) [f32] -> row-major (K, H*hb) [bf16]
//   out[k*N + h*hb + j] = in[(h*K + k)*hb + j],  N = H*hb
// ---------------------------------------------------------------------------
__global__ __launch_bounds__(256)
void permute_head_w(const float* __restrict__ in, __bf16* __restrict__ out,
                    int Kd, int N, int hb)
{
    long i = (long)blockIdx.x * 256 + threadIdx.x;
    if (i >= (long)Kd * N) return;
    int k = (int)(i / N);
    int n = (int)(i - (long)k * N);
    int h = n / hb;
    int j = n - h * hb;
    out[i] = (__bf16)in[(long)(h * Kd + k) * hb + j];
}

// ---------------------------------------------------------------------------
// GEMM: D(M,N) = A(M,K)[bf16] * W(K,N)[bf16 row-major]
// Block = 256 threads (8 waves); block tile 256(M) x 64(N); wave tile 32 x 64.
// Per K-step (32): stage 32x64 W tile transposed in LDS (B-frags become two
// ds_load_b128), two A fragments per wave share four B fragments -> 8 WMMA.
// WGP-scope prefetch of the next W tile.
// Epilogue fuses +bias, +residual(f32), ReLU, f32 and/or bf16 stores.
// ---------------------------------------------------------------------------
__global__ __launch_bounds__(256)
void gemm_bf16_wmma(const __bf16* __restrict__ A,
                    const __bf16* __restrict__ Wb,
                    int M, int N, int K,
                    const float* __restrict__ bias,
                    const float* __restrict__ residual,
                    float*  __restrict__ outF,
                    __bf16* __restrict__ outB,
                    int relu)
{
    __shared__ __bf16 ldsW[64 * 32];   // [n_local][k_local] transposed, 4 KB

    const int tid  = threadIdx.x;
    const int wave = tid >> 5;
    const int lane = tid & 31;
    const int hl   = lane >> 4;        // wave32 half
    const int l16  = lane & 15;

    const int n0 = blockIdx.x * 64;
    const int m0 = blockIdx.y * 256;
    const int mw = m0 + wave * 32;     // this wave's 32 rows

    v8f acc[2][4] = {};

    // A fragment bases (ISA 16-bit A-matrix 16x32 layout)
    const __bf16* Al0 = A + (size_t)(mw + l16) * K + hl * 8;
    const __bf16* Al1 = A + (size_t)(mw + 16 + l16) * K + hl * 8;

    const int kkS  = tid >> 3;         // 0..31 : staged k row
    const int nOff = (tid & 7) * 8;    // 0..56 : staged n chunk

    for (int k0 = 0; k0 < K; k0 += 32) {
        // ---- stage 32x64 bf16 weight tile, transposed into LDS ----
        {
            const __bf16* wp = Wb + (size_t)(k0 + kkS) * N + n0 + nOff;
            uint4 wv = *(const uint4*)wp;          // 8 bf16, one b128 load
            const __bf16* we = (const __bf16*)&wv;
            #pragma unroll
            for (int j = 0; j < 8; ++j)
                ldsW[(nOff + j) * 32 + kkS] = we[j];
        }
        // prefetch next K-step's weight rows (one lane per row, WGP scope)
        if (tid < 32 && (k0 + 32) < K)
            __builtin_prefetch((const void*)(Wb + (size_t)(k0 + 32 + tid) * N + n0), 0, 3);
        __syncthreads();

        // ---- two A fragments (four 16B global loads) ----
        BF16x16 a0, a1;
        a0.u[0] = *(const uint4*)(Al0 + k0);
        a0.u[1] = *(const uint4*)(Al0 + k0 + 16);
        a1.u[0] = *(const uint4*)(Al1 + k0);
        a1.u[1] = *(const uint4*)(Al1 + k0 + 16);

        // ---- 4 B fragments from LDS, each reused by both A fragments ----
        #pragma unroll
        for (int nt = 0; nt < 4; ++nt) {
            BF16x16 bU;
            const __bf16* bp = &ldsW[(nt * 16 + l16) * 32 + hl * 8];
            bU.u[0] = *(const uint4*)(bp);
            bU.u[1] = *(const uint4*)(bp + 16);
            acc[0][nt] = __builtin_amdgcn_wmma_f32_16x16x32_bf16(
                false, a0.v, false, bU.v, (short)0, acc[0][nt], false, false);
            acc[1][nt] = __builtin_amdgcn_wmma_f32_16x16x32_bf16(
                false, a1.v, false, bU.v, (short)0, acc[1][nt], false, false);
        }
        __syncthreads();
    }

    // ---- epilogue: row = mw + sub*16 + 8*hl + r, col = n0 + nt*16 + l16 ----
    #pragma unroll
    for (int sub = 0; sub < 2; ++sub) {
        #pragma unroll
        for (int nt = 0; nt < 4; ++nt) {
            const int col = n0 + nt * 16 + l16;
            const float bv = bias ? bias[col] : 0.0f;
            #pragma unroll
            for (int r = 0; r < 8; ++r) {
                const int row = mw + sub * 16 + hl * 8 + r;
                float v = acc[sub][nt][r] + bv;
                if (residual) v += residual[(long)row * N + col];
                if (relu) v = fmaxf(v, 0.0f);
                const long idx = (long)row * N + col;
                if (outF) outF[idx] = v;
                if (outB) outB[idx] = (__bf16)v;
            }
        }
    }
}

// ---------------------------------------------------------------------------
// Flash attention (bf16 in/out, f32 online-softmax state).
// Block = 128 threads = 4 waves; each wave owns one 16-row Q tile.
// Per 32-column S block:
//   S = Q*K^T (4 WMMA)  ->  online softmax (shfl_xor inside 16-lane halves)
//   V block DMA'd into wave-private LDS via global_load_async_to_lds_b128
//   P -> LDS -> A-layout, O += P*V (4 WMMA)
// ---------------------------------------------------------------------------
__global__ __launch_bounds__(128)
void attn_wmma(const __bf16* __restrict__ Qb, int qStride, int qHeadMul,
               const __bf16* __restrict__ Kb,
               const __bf16* __restrict__ Vb, int kvStride, int kvHeadMul,
               __bf16* __restrict__ Ob, int oStride,
               int T, float scale, int causal)
{
    __shared__ __bf16 ldsP[4 * 16 * 32];   // per-wave 16x32 P tile, 4 KB
    __shared__ __bf16 ldsV[4 * 32 * 64];   // per-wave 32x64 V tile, 16 KB

    const int tid  = threadIdx.x;
    const int wave = tid >> 5;
    const int lane = tid & 31;
    const int hl   = lane >> 4;
    const int l16  = lane & 15;

    const int b = blockIdx.z;
    const int h = blockIdx.y;
    const int qt  = blockIdx.x * 4 + wave;
    const int qr0 = qt * 16;

    const __bf16* q = Qb + (long)b * T * qStride + (long)h * qHeadMul;
    const __bf16* k = Kb + (long)b * T * kvStride + (long)h * kvHeadMul;
    const __bf16* v = Vb + (long)b * T * kvStride + (long)h * kvHeadMul;
    __bf16*       o = Ob + (long)b * T * oStride + h * 64;

    // Q fragments for D=64 (two K=32 steps), ISA A-layout
    BF16x16 aQ[2];
    {
        const __bf16* ql = q + (long)(qr0 + l16) * qStride + hl * 8;
        #pragma unroll
        for (int kk = 0; kk < 2; ++kk) {
            aQ[kk].u[0] = *(const uint4*)(ql + kk * 32);
            aQ[kk].u[1] = *(const uint4*)(ql + kk * 32 + 16);
        }
    }

    float mrow[8], lrow[8];
    #pragma unroll
    for (int r = 0; r < 8; ++r) { mrow[r] = -3.0e38f; lrow[r] = 0.0f; }
    v8f acc[4] = {};

    __bf16* pl = &ldsP[wave * 16 * 32];
    __bf16* vw = &ldsV[wave * 32 * 64];
    const unsigned vwOff = lds_offset(vw);
    const int sEnd = causal ? (qr0 + 16) : T;

    for (int s0 = 0; s0 < sEnd; s0 += 32) {
        // ---- kick off async DMA of the 32x64 V block into wave-private LDS ----
        {
            const char* vbase = (const char*)v + (long)s0 * kvStride * 2;
            #pragma unroll
            for (int i = 0; i < 8; ++i) {
                const int c   = i * 32 + lane;       // 16B chunk id, 0..255
                const int row = c >> 3;
                const int chk = c & 7;
                unsigned lAddr = vwOff + (unsigned)c * 16u;
                const void* g = vbase + (long)row * kvStride * 2 + chk * 16;
                asm volatile("global_load_async_to_lds_b128 %0, %1, off"
                             :: "v"(lAddr), "v"(g) : "memory");
            }
        }

        // ---- scores: S(16x32) = Q(16x64) * K^T, two 16-col tiles ----
        v8f sc[2] = {};
        #pragma unroll
        for (int nt2 = 0; nt2 < 2; ++nt2) {
            const int scol = s0 + nt2 * 16 + l16;
            const __bf16* kl = k + (long)scol * kvStride + hl * 8;
            BF16x16 bK0, bK1;
            bK0.u[0] = *(const uint4*)(kl);      bK0.u[1] = *(const uint4*)(kl + 16);
            bK1.u[0] = *(const uint4*)(kl + 32); bK1.u[1] = *(const uint4*)(kl + 48);
            sc[nt2] = __builtin_amdgcn_wmma_f32_16x16x32_bf16(
                false, aQ[0].v, false, bK0.v, (short)0, sc[nt2], false, false);
            sc[nt2] = __builtin_amdgcn_wmma_f32_16x16x32_bf16(
                false, aQ[1].v, false, bK1.v, (short)0, sc[nt2], false, false);
        }

        // ---- scale + causal mask (C/D layout: row = qr0+8*hl+r) ----
        #pragma unroll
        for (int nt2 = 0; nt2 < 2; ++nt2) {
            const int col = s0 + nt2 * 16 + l16;
            #pragma unroll
            for (int r = 0; r < 8; ++r) {
                float val = sc[nt2][r] * scale;
                if (causal && (col > qr0 + hl * 8 + r)) val = -3.0e38f;
                sc[nt2][r] = val;
            }
        }

        // ---- online softmax (row reductions inside each 16-lane half) ----
        float alpha[8];
        #pragma unroll
        for (int r = 0; r < 8; ++r) {
            float vmax = fmaxf(sc[0][r], sc[1][r]);
            #pragma unroll
            for (int off = 8; off >= 1; off >>= 1)
                vmax = fmaxf(vmax, __shfl_xor(vmax, off, 32));
            const float mnew = fmaxf(mrow[r], vmax);
            alpha[r] = __expf(mrow[r] - mnew);
            mrow[r] = mnew;
            const float p0 = __expf(sc[0][r] - mnew);
            const float p1 = __expf(sc[1][r] - mnew);
            sc[0][r] = p0; sc[1][r] = p1;
            float ps = p0 + p1;
            #pragma unroll
            for (int off = 8; off >= 1; off >>= 1)
                ps += __shfl_xor(ps, off, 32);
            lrow[r] = lrow[r] * alpha[r] + ps;
        }
        #pragma unroll
        for (int nt = 0; nt < 4; ++nt)
            #pragma unroll
            for (int r = 0; r < 8; ++r)
                acc[nt][r] *= alpha[r];

        // ---- P (C/D layout) -> LDS row-major 16x32 -> reload in A-layout ----
        #pragma unroll
        for (int nt2 = 0; nt2 < 2; ++nt2)
            #pragma unroll
            for (int r = 0; r < 8; ++r)
                pl[(hl * 8 + r) * 32 + nt2 * 16 + l16] = (__bf16)sc[nt2][r];

        BF16x16 aP;
        {
            const __bf16* pr = pl + l16 * 32 + hl * 8;
            aP.u[0] = *(const uint4*)(pr);
            aP.u[1] = *(const uint4*)(pr + 16);
        }

        // ---- wait for V DMA, then O(16x64) += P(16x32) * V(32x64) ----
        asm volatile("s_wait_asynccnt 0" ::: "memory");
        #pragma unroll
        for (int nt = 0; nt < 4; ++nt) {
            BF16x16 bV;
            #pragma unroll
            for (int j = 0; j < 16; ++j)
                bV.e[j] = vw[(hl * 8 + j + (j & 8)) * 64 + nt * 16 + l16];
            acc[nt] = __builtin_amdgcn_wmma_f32_16x16x32_bf16(
                false, aP.v, false, bV.v, (short)0, acc[nt], false, false);
        }
    }

    // ---- normalize and store merged bf16 output ----
    #pragma unroll
    for (int r = 0; r < 8; ++r) {
        const float inv = 1.0f / lrow[r];
        const int row = qr0 + hl * 8 + r;
        #pragma unroll
        for (int nt = 0; nt < 4; ++nt)
            o[(long)row * oStride + nt * 16 + l16] = (__bf16)(acc[nt][r] * inv);
    }
}

// ---------------------------------------------------------------------------
// Host-side orchestration
// ---------------------------------------------------------------------------
extern "C" void kernel_launch(void* const* d_in, const int* in_sizes, int n_in,
                              void* d_out, int out_size, void* d_ws, size_t ws_size,
                              hipStream_t stream)
{
    const int B = 4, T = 1024, C = 1024, H = 16, D = 64, FF = 4096;
    const int M = B * T;

    const float* x0     = (const float*)d_in[0];
    const float* enc    = (const float*)d_in[1];
    const float* wqkv   = (const float*)d_in[2];
    const float* wproj  = (const float*)d_in[3];
    const float* bproj  = (const float*)d_in[4];
    const float* cwq    = (const float*)d_in[5];
    const float* cwkv   = (const float*)d_in[6];
    const float* cwproj = (const float*)d_in[7];
    const float* cbproj = (const float*)d_in[8];
    const float* ff1w1  = (const float*)d_in[9];
    const float* ff1b1  = (const float*)d_in[10];
    const float* ff1w2  = (const float*)d_in[11];
    const float* ff1b2  = (const float*)d_in[12];
    const float* ff2w1  = (const float*)d_in[13];
    const float* ff2b1  = (const float*)d_in[14];
    const float* ff2w2  = (const float*)d_in[15];
    const float* ff2b2  = (const float*)d_in[16];
    const float* ln1w = (const float*)d_in[17]; const float* ln1b = (const float*)d_in[18];
    const float* ln2w = (const float*)d_in[19]; const float* ln2b = (const float*)d_in[20];
    const float* ln3w = (const float*)d_in[21]; const float* ln3b = (const float*)d_in[22];
    const float* ln4w = (const float*)d_in[23]; const float* ln4b = (const float*)d_in[24];
    float* out = (float*)d_out;

    // ---- workspace carve-up (~166 MB) ----
    char* ws = (char*)d_ws;
    size_t off = 0;
    auto take = [&](size_t bytes) -> char* {
        char* p = ws + off;
        off += (bytes + 255) & ~(size_t)255;
        return p;
    };
    // activations
    __bf16* h_bf   = (__bf16*)take((size_t)M * C * 2);        // LN outputs (reused 4x)
    __bf16* qkv_bf = (__bf16*)take((size_t)M * 3 * C * 2);    // self qkv; reused for cross q+kv
    __bf16* ao_bf  = (__bf16*)take((size_t)M * C * 2);        // merged attention output
    __bf16* ff_bf  = (__bf16*)take((size_t)M * FF * 2);       // relu hidden (reused 2x)
    __bf16* enc_bf = (__bf16*)take((size_t)M * C * 2);
    float*  xA     = (float*)take((size_t)M * C * 4);
    float*  xB     = (float*)take((size_t)M * C * 4);
    // bf16 weights, row-major (K,N)
    __bf16* wqkv_b  = (__bf16*)take((size_t)C * 3 * C * 2);
    __bf16* wproj_b = (__bf16*)take((size_t)C * C * 2);
    __bf16* cwq_b   = (__bf16*)take((size_t)C * C * 2);
    __bf16* cwkv_b  = (__bf16*)take((size_t)C * 2 * C * 2);
    __bf16* cwproj_b= (__bf16*)take((size_t)C * C * 2);
    __bf16* ff1w1_b = (__bf16*)take((size_t)C * FF * 2);
    __bf16* ff1w2_b = (__bf16*)take((size_t)FF * C * 2);
    __bf16* ff2w1_b = (__bf16*)take((size_t)C * FF * 2);
    __bf16* ff2w2_b = (__bf16*)take((size_t)FF * C * 2);

    __bf16* qx_bf = qkv_bf;                  // cross q  (M x C)
    __bf16* kv_bf = qkv_bf + (size_t)M * C;  // cross kv (M x 2C)

    const dim3 blk128(128), blk256(256);
    auto cdiv = [](long a, long b) { return (unsigned)((a + b - 1) / b); };

    // ===== weight preparation (bf16, row-major) =====
    permute_head_w<<<cdiv((long)C * 3 * C, 256), blk256, 0, stream>>>(wqkv, wqkv_b, C, 3 * C, 3 * D);
    permute_head_w<<<cdiv((long)C * C, 256),     blk256, 0, stream>>>(cwq,  cwq_b,  C, C,     D);
    permute_head_w<<<cdiv((long)C * 2 * C, 256), blk256, 0, stream>>>(cwkv, cwkv_b, C, 2 * C, 2 * D);
    cast_f32_to_bf16<<<cdiv((long)C * C, 256),  blk256, 0, stream>>>(wproj,  wproj_b,  (long)C * C);
    cast_f32_to_bf16<<<cdiv((long)C * C, 256),  blk256, 0, stream>>>(cwproj, cwproj_b, (long)C * C);
    cast_f32_to_bf16<<<cdiv((long)C * FF, 256), blk256, 0, stream>>>(ff1w1, ff1w1_b, (long)C * FF);
    cast_f32_to_bf16<<<cdiv((long)FF * C, 256), blk256, 0, stream>>>(ff1w2, ff1w2_b, (long)FF * C);
    cast_f32_to_bf16<<<cdiv((long)C * FF, 256), blk256, 0, stream>>>(ff2w1, ff2w1_b, (long)C * FF);
    cast_f32_to_bf16<<<cdiv((long)FF * C, 256), blk256, 0, stream>>>(ff2w2, ff2w2_b, (long)FF * C);
    cast_f32_to_bf16<<<cdiv((long)M * C, 256),  blk256, 0, stream>>>(enc, enc_bf, (long)M * C);

    // ===== self-attention =====
    layernorm_bf16<<<M, 256, 0, stream>>>(x0, ln1w, ln1b, h_bf);
    gemm_bf16_wmma<<<dim3((3 * C) / 64, M / 256), blk256, 0, stream>>>(
        h_bf, wqkv_b, M, 3 * C, C, nullptr, nullptr, nullptr, qkv_bf, 0);
    attn_wmma<<<dim3(T / 64, H, B), blk128, 0, stream>>>(
        qkv_bf, 3 * C, 3 * D,
        qkv_bf + D, qkv_bf + 2 * D, 3 * C, 3 * D,
        ao_bf, C, T, 0.125f, /*causal=*/1);
    gemm_bf16_wmma<<<dim3(C / 64, M / 256), blk256, 0, stream>>>(
        ao_bf, wproj_b, M, C, C, bproj, x0, xA, nullptr, 0);

    // ===== feed-forward 1 =====
    layernorm_bf16<<<M, 256, 0, stream>>>(xA, ln2w, ln2b, h_bf);
    gemm_bf16_wmma<<<dim3(FF / 64, M / 256), blk256, 0, stream>>>(
        h_bf, ff1w1_b, M, FF, C, ff1b1, nullptr, nullptr, ff_bf, /*relu=*/1);
    gemm_bf16_wmma<<<dim3(C / 64, M / 256), blk256, 0, stream>>>(
        ff_bf, ff1w2_b, M, C, FF, ff1b2, xA, xB, nullptr, 0);

    // ===== cross-attention =====
    layernorm_bf16<<<M, 256, 0, stream>>>(xB, ln3w, ln3b, h_bf);
    gemm_bf16_wmma<<<dim3(C / 64, M / 256), blk256, 0, stream>>>(
        h_bf, cwq_b, M, C, C, nullptr, nullptr, nullptr, qx_bf, 0);
    gemm_bf16_wmma<<<dim3((2 * C) / 64, M / 256), blk256, 0, stream>>>(
        enc_bf, cwkv_b, M, 2 * C, C, nullptr, nullptr, nullptr, kv_bf, 0);
    attn_wmma<<<dim3(T / 64, H, B), blk128, 0, stream>>>(
        qx_bf, C, D,
        kv_bf, kv_bf + D, 2 * C, 2 * D,
        ao_bf, C, T, 0.125f, /*causal=*/0);
    gemm_bf16_wmma<<<dim3(C / 64, M / 256), blk256, 0, stream>>>(
        ao_bf, cwproj_b, M, C, C, cbproj, xB, xA, nullptr, 0);

    // ===== feed-forward 2 =====
    layernorm_bf16<<<M, 256, 0, stream>>>(xA, ln4w, ln4b, h_bf);
    gemm_bf16_wmma<<<dim3(FF / 64, M / 256), blk256, 0, stream>>>(
        h_bf, ff2w1_b, M, FF, C, ff2b1, nullptr, nullptr, ff_bf, /*relu=*/1);
    gemm_bf16_wmma<<<dim3(C / 64, M / 256), blk256, 0, stream>>>(
        ff_bf, ff2w2_b, M, C, FF, ff2b2, xA, out, nullptr, 0);
}
